// Net_61272003445332
// MI455X (gfx1250) — matrix operand order repaired and checked by reading, not compiled, not run on hardware
//
#include <hip/hip_runtime.h>
#include <hip/hip_bf16.h>
#include <hip/hip_fp16.h>

// ---------------------------------------------------------------------------
// GraphSAGE + LSTM aggregator for MI455X (gfx1250, wave32, WMMA).
// N=20000 nodes, K=16 neighbors, dims 64 -> 128 -> 128 -> 32.
// Compute-bound (~190 GFLOP); features fit in L2; weights staged f16 in LDS.
// All GEMMs via v_wmma_f32_16x16x32_f16 (f16 in, f32 accumulate).
// Gate nonlinearities use HW tanh (v_tanh_f32) / v_rcp_f32 fast paths.
// ---------------------------------------------------------------------------

#define NNODES 20000
#define KNEIGH 16

typedef _Float16 v16h __attribute__((ext_vector_type(16)));
typedef _Float16 v8h  __attribute__((ext_vector_type(8)));
typedef float    v8f  __attribute__((ext_vector_type(8)));

#if __has_builtin(__builtin_amdgcn_tanhf)
__device__ __forceinline__ float tanh_(float x) {
    return __builtin_amdgcn_tanhf(x);
}
#else
__device__ __forceinline__ float tanh_(float x) {
    x = fminf(fmaxf(x, -15.0f), 15.0f);
    float e = __expf(2.0f * x);
    return (e - 1.0f) * __builtin_amdgcn_rcpf(e + 1.0f);
}
#endif
__device__ __forceinline__ float sigm_(float x) {
    // sigmoid(x) = 0.5 * (1 + tanh(x/2)) -> one transcendental op
    return fmaf(0.5f, tanh_(0.5f * x), 0.5f);
}

// D = A(16x32) * B(32x16) + C, f16 inputs, f32 accum.
__device__ __forceinline__ v8f wmma16(v16h a, v16h b, v8f c) {
    return __builtin_amdgcn_wmma_f32_16x16x32_f16(
        /*neg_a=*/false, a, /*neg_b=*/false, b,
        /*c_mod=*/(short)0, c, /*reuse_a=*/false, /*reuse_b=*/false);
}

// A-matrix fragment (16x32 f16, MxK). Lane m=lane&15 is row m.
// lanes 0-15: elements 0..7 = K 0..7, 8..15 = K 16..23
// lanes 16-31: elements 0..7 = K 8..15, 8..15 = K 24..31
template <typename PT>
__device__ __forceinline__ v16h load_afrag(PT row, int kc, int lane) {
    const int hi = (lane >> 4) & 1;
    v8h lo = *(const v8h*)(row + kc * 32 + hi * 8);
    v8h hh = *(const v8h*)(row + kc * 32 + hi * 8 + 16);
    v16h a;
#pragma unroll
    for (int i = 0; i < 8; i++) { a[i] = lo[i]; a[i + 8] = hh[i]; }
    return a;
}

// B-matrix fragment (32x16 f16, KxN) from a TRANSPOSED weight WT[k][rowStride].
// Lane L holds column col0+(L&15); elements 0..7 = K 8h..8h+7, 8..15 = K 16+8h..23+8h (h=L>>4).
template <typename PT>
__device__ __forceinline__ v16h load_bfrag(PT WT, int rowStride, int k0,
                                           int col0, int lane) {
    const int hi = (lane >> 4) & 1;
    PT p = WT + (k0 + hi * 8) * rowStride + col0 + (lane & 15);
    v16h b;
#pragma unroll
    for (int i = 0; i < 8; i++) { b[i] = p[i * rowStride]; b[i + 8] = p[(i + 16) * rowStride]; }
    return b;
}

// feat0 = cat(deg/N, p) converted to f16.  (N x 64)
__global__ void prep_feat_kernel(const float* __restrict__ p,
                                 _Float16* __restrict__ feat0) {
    int t = blockIdx.x * blockDim.x + threadIdx.x;
    if (t >= NNODES * 64) return;
    int node = t >> 6;
    int c = t & 63;
    float v = (c == 0) ? ((float)KNEIGH / (float)NNODES)
                       : p[node * 63 + (c - 1)];
    feat0[t] = (_Float16)v;
}

// One wave processes one 16-node tile through the K-step LSTM aggregator
// plus the fc_self + fc_neigh output GEMM.
template <int DIN, int DOUT, int NW, bool RELU, bool F16OUT>
__global__ __launch_bounds__(NW * 32, 1)
void lstm_sage_layer(const _Float16* __restrict__ featIn,     // N x DIN
                     const int* __restrict__ neighIdx,        // N x K
                     const float* __restrict__ Wih,           // 4DIN x DIN
                     const float* __restrict__ Whh,           // 4DIN x DIN
                     const float* __restrict__ bih,
                     const float* __restrict__ bhh,
                     const float* __restrict__ Wself,         // DOUT x DIN
                     const float* __restrict__ Wneigh,        // DOUT x DIN
                     const float* __restrict__ bOut,          // DOUT
                     _Float16* __restrict__ outH,             // N x DOUT (f16)
                     float* __restrict__ outF)                // N x DOUT (f32)
{
    constexpr int KC = DIN / 32;   // K-chunks per GEMM
    constexpr int NJ = DIN / 16;   // hidden column tiles

    extern __shared__ __align__(16) char smemRaw[];
    _Float16* WihT  = (_Float16*)smemRaw;           // [DIN][4*DIN] transposed
    _Float16* WhhT  = WihT + 4 * DIN * DIN;         // [DIN][4*DIN] transposed
    float*    bias4 = (float*)(WihT + 8 * DIN * DIN);   // [4*DIN] bih+bhh
    _Float16* hstAll = (_Float16*)(bias4 + 4 * DIN);    // [NW][16][DIN]

    const int tid = threadIdx.x;
    const int nth = NW * 32;

    // Stage transposed f16 LSTM weights + summed bias into LDS (once).
    for (int e = tid; e < 4 * DIN * DIN; e += nth) {
        int r = e / DIN, k = e - r * DIN;
        WihT[k * (4 * DIN) + r] = (_Float16)Wih[e];
        WhhT[k * (4 * DIN) + r] = (_Float16)Whh[e];
    }
    for (int e = tid; e < 4 * DIN; e += nth) bias4[e] = bih[e] + bhh[e];
    __syncthreads();

    const int wave = tid >> 5;
    const int lane = tid & 31;
    const int m    = lane & 15;          // tile row (A) / tile col (D)
    const int hi8  = (lane >> 4) << 3;   // D-layout row offset
    const int tileBase = (blockIdx.x * NW + wave) * 16;
    const bool active = (tileBase < NNODES);
    _Float16* hst = hstAll + wave * 16 * DIN;

    v8f  cst[NJ];     // cell state, D layout
    v16h Ah[KC];      // hidden state as A fragments
#pragma unroll
    for (int j = 0; j < NJ; j++) {
#pragma unroll
        for (int i = 0; i < 8; i++) cst[j][i] = 0.0f;
    }
#pragma unroll
    for (int kc = 0; kc < KC; kc++) {
#pragma unroll
        for (int i = 0; i < 16; i++) Ah[kc][i] = (_Float16)0.0f;
    }

    if (active) {
        for (int t = 0; t < KNEIGH; t++) {
            // Gather neighbor-t features for this tile's 16 rows.
            int idx = neighIdx[(tileBase + m) * KNEIGH + t];
            const _Float16* xrow = featIn + (long)idx * DIN;
            v16h Ax[KC];
#pragma unroll
            for (int kc = 0; kc < KC; kc++) Ax[kc] = load_afrag(xrow, kc, lane);

#pragma unroll
            for (int j = 0; j < NJ; j++) {
                v8f g4[4];
#pragma unroll
                for (int g = 0; g < 4; g++) {        // gate order i,f,g,o
                    float bv = bias4[g * DIN + j * 16 + m];
                    v8f acc;
#pragma unroll
                    for (int i = 0; i < 8; i++) acc[i] = bv;
#pragma unroll
                    for (int kc = 0; kc < KC; kc++) {
                        acc = wmma16(Ax[kc],
                                     load_bfrag(WihT, 4 * DIN, kc * 32,
                                                g * DIN + j * 16, lane), acc);
                        acc = wmma16(Ah[kc],
                                     load_bfrag(WhhT, 4 * DIN, kc * 32,
                                                g * DIN + j * 16, lane), acc);
                    }
                    g4[g] = acc;
                }
                // Elementwise LSTM cell update in D layout.
#pragma unroll
                for (int i = 0; i < 8; i++) {
                    float iv = sigm_(g4[0][i]);
                    float fv = sigm_(g4[1][i]);
                    float gv = tanh_(g4[2][i]);
                    float ov = sigm_(g4[3][i]);
                    float cv = fmaf(fv, cst[j][i], iv * gv);
                    cst[j][i] = cv;
                    // D layout -> LDS staging (row i+hi8, col j*16+m)
                    hst[(i + hi8) * DIN + j * 16 + m] =
                        (_Float16)(ov * tanh_(cv));
                }
            }
            // Reload h as A fragments (in-wave LDS transpose; DS in-order).
#pragma unroll
            for (int kc = 0; kc < KC; kc++)
                Ah[kc] = load_afrag(hst + m * DIN, kc, lane);
        }
    }
    __syncthreads();

    // Reuse the (dead) LSTM weight region for transposed output weights.
    _Float16* WsT = (_Float16*)smemRaw;      // [DIN][DOUT]
    _Float16* WnT = WsT + DIN * DOUT;        // [DIN][DOUT]
    for (int e = tid; e < DIN * DOUT; e += nth) {
        int r = e / DIN, k = e - r * DIN;
        WsT[k * DOUT + r] = (_Float16)Wself[e];
        WnT[k * DOUT + r] = (_Float16)Wneigh[e];
    }
    __syncthreads();

    if (active) {
        const _Float16* frow = featIn + (long)(tileBase + m) * DIN;
        v16h Af[KC];
#pragma unroll
        for (int kc = 0; kc < KC; kc++) Af[kc] = load_afrag(frow, kc, lane);

#pragma unroll
        for (int jo = 0; jo < DOUT / 16; jo++) {
            float bv = bOut[jo * 16 + m];
            v8f acc;
#pragma unroll
            for (int i = 0; i < 8; i++) acc[i] = bv;
#pragma unroll
            for (int kc = 0; kc < KC; kc++) {
                acc = wmma16(Af[kc],
                             load_bfrag(WsT, DOUT, kc * 32, jo * 16, lane), acc);
                acc = wmma16(Ah[kc],
                             load_bfrag(WnT, DOUT, kc * 32, jo * 16, lane), acc);
            }
#pragma unroll
            for (int i = 0; i < 8; i++) {
                float v = acc[i];
                if (RELU) v = fmaxf(v, 0.0f);
                long r = tileBase + i + hi8;
                if (F16OUT) outH[r * DOUT + jo * 16 + m] = (_Float16)v;
                else        outF[r * DOUT + jo * 16 + m] = v;
            }
        }
    }
}

static inline size_t layer_smem_bytes(int din, int nw) {
    return (size_t)8 * din * din * 2     // WihT + WhhT (f16)
         + (size_t)4 * din * 4           // bias4 (f32)
         + (size_t)nw * 16 * din * 2;    // h staging (f16)
}

extern "C" void kernel_launch(void* const* d_in, const int* in_sizes, int n_in,
                              void* d_out, int out_size, void* d_ws,
                              size_t ws_size, hipStream_t stream) {
    (void)in_sizes; (void)n_in; (void)out_size; (void)ws_size;

    const float* p   = (const float*)d_in[0];
    const int* nidx  = (const int*)d_in[1];
    const float* L0[7]; const float* L1[7]; const float* L2[7];
    for (int i = 0; i < 7; i++) {
        L0[i] = (const float*)d_in[2 + i];
        L1[i] = (const float*)d_in[9 + i];
        L2[i] = (const float*)d_in[16 + i];
    }

    _Float16* buf0 = (_Float16*)d_ws;                   // N x 64
    _Float16* buf1 = buf0 + (size_t)NNODES * 64;        // N x 128
    _Float16* buf2 = buf1 + (size_t)NNODES * 128;       // N x 128
    float* out = (float*)d_out;

    {
        int tot = NNODES * 64;
        prep_feat_kernel<<<(tot + 255) / 256, 256, 0, stream>>>(p, buf0);
    }

    constexpr int NW = 8;                 // 8 waves = 8 node-tiles per block
    const int nTiles  = NNODES / 16;      // 1250 (exact)
    const int nBlocks = (nTiles + NW - 1) / NW;

    // Layer 0: 64 -> 128, ReLU, f16 out
    {
        size_t smem = layer_smem_bytes(64, NW);
        hipFuncSetAttribute(
            reinterpret_cast<const void*>(&lstm_sage_layer<64, 128, NW, true, true>),
            hipFuncAttributeMaxDynamicSharedMemorySize, (int)smem);
        lstm_sage_layer<64, 128, NW, true, true><<<nBlocks, NW * 32, smem, stream>>>(
            buf0, nidx, L0[0], L0[1], L0[2], L0[3], L0[4], L0[5], L0[6],
            buf1, nullptr);
    }
    // Layer 1: 128 -> 128, ReLU, f16 out
    {
        size_t smem = layer_smem_bytes(128, NW);
        hipFuncSetAttribute(
            reinterpret_cast<const void*>(&lstm_sage_layer<128, 128, NW, true, true>),
            hipFuncAttributeMaxDynamicSharedMemorySize, (int)smem);
        lstm_sage_layer<128, 128, NW, true, true><<<nBlocks, NW * 32, smem, stream>>>(
            buf1, nidx, L1[0], L1[1], L1[2], L1[3], L1[4], L1[5], L1[6],
            buf2, nullptr);
    }
    // Layer 2: 128 -> 32, no ReLU, f32 out to d_out
    {
        size_t smem = layer_smem_bytes(128, NW);
        hipFuncSetAttribute(
            reinterpret_cast<const void*>(&lstm_sage_layer<128, 32, NW, false, false>),
            hipFuncAttributeMaxDynamicSharedMemorySize, (int)smem);
        lstm_sage_layer<128, 32, NW, false, false><<<nBlocks, NW * 32, smem, stream>>>(
            buf2, nidx, L2[0], L2[1], L2[2], L2[3], L2[4], L2[5], L2[6],
            nullptr, out);
    }
}
